// CNN_7181185319157
// MI455X (gfx1250) — matrix operand order repaired and checked by reading, not compiled, not run on hardware
//
#include <hip/hip_runtime.h>

typedef __attribute__((ext_vector_type(16))) __bf16 v16bf;
typedef __attribute__((ext_vector_type(8)))  float  v8f;

union AFrag { uint4 q[2]; v16bf v; };
union BFrag { unsigned short u[16]; v16bf v; };

__device__ __forceinline__ unsigned short f2bf(float f) {
    union { float f; unsigned int u; } c; c.f = f;
    unsigned int x = c.u;
    return (unsigned short)((x + 0x7FFFu + ((x >> 16) & 1u)) >> 16);   // RNE
}
__device__ __forceinline__ unsigned int f2bf2(float lo, float hi) {
    union { float f; unsigned int u; } a, b; a.f = lo; b.f = hi;
    unsigned int rl = (a.u + 0x7FFFu + ((a.u >> 16) & 1u)) >> 16;
    unsigned int rh = (b.u + 0x7FFFu + ((b.u >> 16) & 1u)) & 0xFFFF0000u;
    return rl | rh;
}

// Sizes fixed by the reference: V=50000, D=300, H=64, S=512, B=256, PAD=1
// Fused: gather -> bf16 WMMA GEMM (K=600 padded to 608, N=128=[W0|W1]) ->
//        conv[t] = g[t,0:64] + g[t+1,64:128] + bias -> masked max over t -> head.
__global__ __launch_bounds__(256) void cnn_fused(
    const int*   __restrict__ sent,   // [512][256]
    const float* __restrict__ E,      // [50000][300]
    const float* __restrict__ U,      // [50000][300]
    const float* __restrict__ convw,  // [64][600][2]
    const float* __restrict__ convb,  // [64]
    const float* __restrict__ outw,   // [2][64]
    const float* __restrict__ outb,   // [2]
    float*       __restrict__ out)    // [256][2]
{
    extern __shared__ char smem[];
    float*          g      = (float*)smem;                              // 512*128*4   = 262144 B
    unsigned int*   ast32  = (unsigned int*)(smem + 262144);            // 2*16*304 u32=  38912 B
    unsigned short* ast16  = (unsigned short*)(smem + 262144);          //   (same memory)
    int*            ldsTok = (int*)(smem + 262144 + 38912);             // 512*4       =   2048 B
    float*          red    = (float*)(smem + 262144 + 38912 + 2048);    // 4*64*4      =   1024 B

    const int tid  = threadIdx.x;
    const int b    = blockIdx.x;
    const int lane = tid & 31;
    const int wave = tid >> 5;       // 0..7 : owns output cols [16w,16w+16)
    const int half = lane >> 4;      // K-half selector for A/B fragments
    const int nloc = lane & 15;      // col within N-tile (B/C/D), row within M-tile (A)

    // ---- token ids for whole sentence ----
    for (int t = tid; t < 512; t += 256) ldsTok[t] = sent[t * 256 + b];

    // ---- tile-invariant gather mapping: pair index -> (token-in-tile, feature col) ----
    // pair idx2 = tid + 256*i covers 16 tokens * 304 pairs; LDS u32 offset == idx2.
    int ptok[19], pcol[19];
    #pragma unroll
    for (int i = 0; i < 19; ++i) {
        int idx2 = tid + (i << 8);
        int token = idx2 / 304;
        ptok[i] = token;
        pcol[i] = (idx2 - token * 304) << 1;      // even feature column c (pairs c,c+1)
    }

    // ---- B-matrix fragments (weights) held in registers for the whole kernel ----
    // B[k][n]: n<64 -> conv_w[n][k][0], n>=64 -> conv_w[n-64][k][1]
    // per-VGPR K layout (32x16 bf16 B): VGPR v holds K = 2(v&3) + 16*(v>>2) + 8*half (+chunk*32)
    v16bf bfrag[19];
    {
        const int n  = (wave << 4) + nloc;
        const int h  = n & 63;
        const int kk = n >> 6;
        #pragma unroll
        for (int kc = 0; kc < 19; ++kc) {
            BFrag bf;
            #pragma unroll
            for (int v = 0; v < 8; ++v) {
                int k = kc * 32 + ((v & 3) << 1) + ((v >> 2) << 4) + half * 8;
                float f0 = (k     < 600) ? convw[h * 1200 + k * 2 + kk]       : 0.f;
                float f1 = (k + 1 < 600) ? convw[h * 1200 + (k + 1) * 2 + kk] : 0.f;
                bf.u[2 * v]     = f2bf(f0);
                bf.u[2 * v + 1] = f2bf(f1);
            }
            bfrag[kc] = bf.v;
        }
    }
    __syncthreads();   // ldsTok ready

    // ---- stage M-tile 0 into buffer 0 (16 tokens x 608 bf16, cols 600..607 zero) ----
    #pragma unroll
    for (int i = 0; i < 19; ++i) {
        int c = pcol[i];
        float2 v = make_float2(0.f, 0.f);
        if (c < 600) {
            int id = ldsTok[ptok[i]];
            const float* base = (c < 300) ? (E + id * 300 + c) : (U + id * 300 + (c - 300));
            v = *(const float2*)base;              // 8B-aligned: rows are 300 floats, c even
        }
        ast32[tid + (i << 8)] = f2bf2(v.x, v.y);
    }
    __syncthreads();

    const int mrow = nloc;   // A-fragment row (M) within tile

    for (int mt = 0; mt < 32; ++mt) {
        // --- issue global gather for next tile BEFORE compute (latency overlap) ---
        float2 pf[19];
        const bool hasNext = (mt + 1) < 32;
        if (hasNext) {
            #pragma unroll
            for (int i = 0; i < 19; ++i) {
                int c = pcol[i];
                float2 v = make_float2(0.f, 0.f);
                if (c < 600) {
                    int id = ldsTok[((mt + 1) << 4) + ptok[i]];
                    const float* base = (c < 300) ? (E + id * 300 + c)
                                                  : (U + id * 300 + (c - 300));
                    v = *(const float2*)base;
                }
                pf[i] = v;
            }
        }

        // --- GEMM: 19 K-chunks, software-pipelined A fragments, two acc chains ---
        const unsigned short* pbase = ast16 + (mt & 1) * 9728 + mrow * 608 + half * 8;
        v8f acc0 = {0.f,0.f,0.f,0.f,0.f,0.f,0.f,0.f};
        v8f acc1 = {0.f,0.f,0.f,0.f,0.f,0.f,0.f,0.f};
        AFrag afa, afb;
        afa.q[0] = *(const uint4*)(pbase);
        afa.q[1] = *(const uint4*)(pbase + 16);
        #pragma unroll
        for (int kc = 0; kc < 19; ++kc) {
            if (kc + 1 < 19) {                     // prefetch next A fragment
                const unsigned short* p = pbase + (kc + 1) * 32;
                if (kc & 1) { afa.q[0] = *(const uint4*)(p); afa.q[1] = *(const uint4*)(p + 16); }
                else        { afb.q[0] = *(const uint4*)(p); afb.q[1] = *(const uint4*)(p + 16); }
            }
            if (kc & 1)
                acc1 = __builtin_amdgcn_wmma_f32_16x16x32_bf16(
                           false, afb.v, false, bfrag[kc], (short)0, acc1, false, false);
            else
                acc0 = __builtin_amdgcn_wmma_f32_16x16x32_bf16(
                           false, afa.v, false, bfrag[kc], (short)0, acc0, false, false);
        }

        // --- commit prefetched gather (convert here so no load-wait blocks the GEMM) ---
        if (hasNext) {
            unsigned int* nbuf = ast32 + ((mt + 1) & 1) * 4864;
            #pragma unroll
            for (int i = 0; i < 19; ++i)
                nbuf[tid + (i << 8)] = f2bf2(pf[i].x, pf[i].y);
        }

        // --- store C tile to g[512][128] in LDS ---
        #pragma unroll
        for (int r = 0; r < 8; ++r) {
            int M = r + (half << 3);
            g[(((mt << 4) + M) << 7) + (wave << 4) + nloc] = acc0[r] + acc1[r];
        }
        __syncthreads();
    }

    // ---- phase 2: conv-add + masked max over time ----
    {
        const int h   = tid & 63;
        const int grp = tid >> 6;          // 4 groups strided over t
        const float cb = convb[h];
        float mx = -__builtin_huge_valf();
        for (int t = grp; t < 511; t += 4) {
            if (ldsTok[t] != 1) {          // t < length  <=>  tok[t] != PAD
                float v = g[(t << 7) + h] + g[((t + 1) << 7) + 64 + h] + cb;
                mx = fmaxf(mx, v);
            }
        }
        red[(grp << 6) + h] = mx;
    }
    __syncthreads();
    if (tid < 64) {
        float p = fmaxf(fmaxf(red[tid], red[64 + tid]),
                        fmaxf(red[128 + tid], red[192 + tid]));
        red[tid] = p;                      // pooled[h]
    }
    __syncthreads();

    // ---- head: scores[b, j] = pooled . out_w[j,:] + out_b[j] ----
    if (tid < 2) {
        float s = outb[tid];
        #pragma unroll 8
        for (int hh = 0; hh < 64; ++hh) s += red[hh] * outw[tid * 64 + hh];
        out[b * 2 + tid] = s;
    }
}

extern "C" void kernel_launch(void* const* d_in, const int* in_sizes, int n_in,
                              void* d_out, int out_size, void* d_ws, size_t ws_size,
                              hipStream_t stream) {
    (void)in_sizes; (void)n_in; (void)out_size; (void)d_ws; (void)ws_size;
    const int*   sent = (const int*)  d_in[0];
    const float* E    = (const float*)d_in[1];
    const float* U    = (const float*)d_in[2];
    const float* cw   = (const float*)d_in[3];
    const float* cb   = (const float*)d_in[4];
    const float* ow   = (const float*)d_in[5];
    const float* ob   = (const float*)d_in[6];
    float* out = (float*)d_out;

    const size_t shmem = 262144 + 38912 + 2048 + 1024;   // 304128 B < 320 KB LDS
    cnn_fused<<<dim3(256), dim3(256), shmem, stream>>>(sent, E, U, cw, cb, ow, ob, out);
}
